// SubMAttention3d_2972117369404
// MI455X (gfx1250) — matrix operand order repaired and checked by reading, not compiled, not run on hardware
//
#include <hip/hip_runtime.h>
#include <hip/hip_bf16.h>
#include <math.h>

#define NVOX 100000
#define CCH  64
#define KNB  32
#define NH   4
#define HDIM 16
#define FF   256
#define OCH  64
#define BN_EPS 1e-5f

typedef __attribute__((ext_vector_type(16))) __bf16 v16bf;
typedef __attribute__((ext_vector_type(8)))  float  v8f;

// ---------------- WMMA fragment helpers (CDNA5 16x16x32 bf16) ----------------
// A (16x32, bf16, row-major source): lane L holds row M=L&15; K-base=(L>>4)*8;
// VGPRs 0..3 -> K = base+0..7, VGPRs 4..7 -> K = base+16..23 (ISA 7.12.2).
__device__ __forceinline__ v16bf frag_a_lds(const __bf16* tile, int ld) {
  const int lane = threadIdx.x & 31;
  const int m  = lane & 15;
  const int kb = (lane >> 4) << 3;
  v16bf a;
#pragma unroll
  for (int j = 0; j < 16; ++j) {
    const int k = kb + j + ((j >> 3) << 3);   // j<8: kb+j ; j>=8: kb+16+(j-8)
    a[j] = tile[m * ld + k];
  }
  return a;
}

// Pre-packed B fragment: one aligned 32B vector load per lane.
__device__ __forceinline__ v16bf frag_b_packed(const __bf16* __restrict__ pk,
                                               int frag) {
  const int lane = threadIdx.x & 31;
  return *(const v16bf*)(pk + ((size_t)frag * 32 + lane) * 16);
}

__device__ __forceinline__ v8f wmma_bf16(v16bf a, v16bf b, v8f c) {
  return __builtin_amdgcn_wmma_f32_16x16x32_bf16(false, a, false, b,
                                                 (short)0, c, false, false);
}

// ---------------- Weight pre-pack: W (row-major f32, D = X @ W.T) ------------
// Fragment (ct, ks): B[k][n] = W[n*ldw + k], n in [ct*16, ct*16+16),
// k in [ks*32, ks*32+32). Lane L holds N=(L&15), K-base=(L>>4)*8; j<8: base+j, j>=8: base+16+(j-8).
// dst layout: dst[(frag*32 + lane)*16 + j], frag = ct*kTiles + ks.
__global__ __launch_bounds__(32) void prepack_kernel(
    const float* __restrict__ W, int ldw, int kTiles, __bf16* __restrict__ dst) {
  const int frag = blockIdx.x;
  const int ct = frag / kTiles, ks = frag % kTiles;
  const int lane = threadIdx.x;
  const int nn = ct * 16 + (lane & 15);
  const int kb = ks * 32 + ((lane >> 4) << 3);
  __bf16* out = dst + ((size_t)frag * 32 + lane) * 16;
#pragma unroll
  for (int j = 0; j < 16; ++j)
    out[j] = (__bf16)W[(size_t)nn * ldw + kb + j + ((j >> 3) << 3)];
}

// ------------------------------- Kernel 1: Q projection ----------------------
// q = (vf + relu(coords @ q_pos_w.T + q_pos_b)) @ wq.T + bq   -> bufA [N,64]
__global__ __launch_bounds__(32) void qproj_kernel(
    const float* __restrict__ vf, const float* __restrict__ vc,
    const __bf16* __restrict__ pwq, const float* __restrict__ bq,
    const float* __restrict__ qpw, const float* __restrict__ qpb,
    float* __restrict__ qout) {
  __shared__ __bf16 qa[16][CCH];
  const int tid = threadIdx.x;
  const int m0  = blockIdx.x * 16;
  for (int e = tid; e < 16 * CCH; e += 32) {
    const int r = e >> 6, c = e & 63;
    const int row = m0 + r;
    const float x = vc[row * 3 + 0], y = vc[row * 3 + 1], z = vc[row * 3 + 2];
    float p = qpw[c * 3 + 0] * x + qpw[c * 3 + 1] * y + qpw[c * 3 + 2] * z + qpb[c];
    qa[r][c] = (__bf16)(vf[(size_t)row * CCH + c] + fmaxf(p, 0.f));
  }
  __syncthreads();
  const int lane = tid & 31;
#pragma unroll
  for (int ct = 0; ct < 4; ++ct) {
    v8f acc = {};
#pragma unroll
    for (int ks = 0; ks < 2; ++ks) {
      v16bf a = frag_a_lds(&qa[0][0] + ks * 32, CCH);
      v16bf b = frag_b_packed(pwq, ct * 2 + ks);
      acc = wmma_bf16(a, b, acc);
    }
    const int col = ct * 16 + (lane & 15);
#pragma unroll
    for (int r = 0; r < 8; ++r) {
      const int row = m0 + ((lane >> 4) << 3) + r;
      qout[(size_t)row * CCH + col] = acc[r] + bq[col];
    }
  }
}

// ------------------------- Kernel 2: fused neighbor attention ----------------
// One workgroup (64 threads = 2 waves) per voxel. Gathered keys stay in LDS;
// wave0 does the K projection, wave1 the V projection (WMMA each), then
// masked softmax over K=32, context, wo matvec, residual -> y1.
__global__ __launch_bounds__(64) void attn_kernel(
    const float* __restrict__ vf, const float* __restrict__ vc,
    const int* __restrict__ kidx, const unsigned char* __restrict__ kmask,
    const __bf16* __restrict__ pwk, const float* __restrict__ bk,
    const __bf16* __restrict__ pwv, const float* __restrict__ bv,
    const float* __restrict__ wo, const float* __restrict__ bo,
    const float* __restrict__ kpw, const float* __restrict__ kpb,
    const float* __restrict__ qin, float* __restrict__ y1) {
  __shared__ __bf16 kf[KNB][CCH];     // gathered keys + pos (bf16)
  __shared__ float  kp[KNB][CCH];     // K projection (f32)
  __shared__ float  vp[KNB][CCH];     // V projection (f32)
  __shared__ float  qs[CCH];          // this voxel's q row
  __shared__ float  sc[NH][KNB];
  __shared__ float  att[NH][KNB];
  __shared__ float  ctxs[CCH];

  const int n   = blockIdx.x;
  const int tid = threadIdx.x;        // == channel 0..63
  const float cx = vc[n * 3 + 0], cy = vc[n * 3 + 1], cz = vc[n * 3 + 2];
  const float w0 = kpw[tid * 3 + 0], w1 = kpw[tid * 3 + 1], w2 = kpw[tid * 3 + 2];
  const float pb = kpb[tid];
  qs[tid] = qin[(size_t)n * CCH + tid];

  for (int j = 0; j < KNB; ++j) {
    const int g = kidx[n * KNB + j];                       // uniform -> scalar
    if (j + 1 < KNB)
      __builtin_prefetch(vf + (size_t)kidx[n * KNB + j + 1] * CCH, 0, 0);
    const float rx = vc[g * 3 + 0] - cx, ry = vc[g * 3 + 1] - cy,
                rz = vc[g * 3 + 2] - cz;
    const float p = fmaxf(w0 * rx + w1 * ry + w2 * rz + pb, 0.f);
    kf[j][tid] = (__bf16)(vf[(size_t)g * CCH + tid] + p);
  }
  __syncthreads();

  // K / V projections: [32,64] @ [64,64] per wave, 16 WMMAs each
  const int w    = tid >> 5;
  const int lane = tid & 31;
  const __bf16* PW = w ? pwv : pwk;
  const float*  Bp = w ? bv : bk;
  float (*outp)[CCH] = w ? vp : kp;
#pragma unroll
  for (int ct = 0; ct < 4; ++ct) {
#pragma unroll
    for (int rt = 0; rt < 2; ++rt) {
      v8f acc = {};
#pragma unroll
      for (int ks = 0; ks < 2; ++ks) {
        v16bf a = frag_a_lds(&kf[rt * 16][0] + ks * 32, CCH);
        v16bf b = frag_b_packed(PW, ct * 2 + ks);
        acc = wmma_bf16(a, b, acc);
      }
      const int col = ct * 16 + (lane & 15);
#pragma unroll
      for (int r = 0; r < 8; ++r) {
        const int row = rt * 16 + ((lane >> 4) << 3) + r;
        outp[row][col] = acc[r] + Bp[col];
      }
    }
  }
  __syncthreads();

  // scores: s[h][j] = (q_h . k_jh) / sqrt(16), masked
  {
    const int h = tid >> 4, j0 = tid & 15;
#pragma unroll
    for (int half = 0; half < 2; ++half) {
      const int j = j0 + half * 16;
      float s = 0.f;
#pragma unroll
      for (int d = 0; d < HDIM; ++d)
        s += qs[h * HDIM + d] * kp[j][h * HDIM + d];
      s *= 0.25f;
      if (kmask[(size_t)n * KNB + j]) s = -1e30f;
      sc[h][j] = s;
    }
  }
  __syncthreads();
  if (tid < NH) {
    float m = -1e30f;
    for (int j = 0; j < KNB; ++j) m = fmaxf(m, sc[tid][j]);
    float sum = 0.f;
    for (int j = 0; j < KNB; ++j) {
      const float e = __expf(sc[tid][j] - m);
      att[tid][j] = e; sum += e;
    }
    const float inv = 1.f / sum;
    for (int j = 0; j < KNB; ++j) att[tid][j] *= inv;
  }
  __syncthreads();
  {
    const int h = tid >> 4;
    float c = 0.f;
    for (int j = 0; j < KNB; ++j) c += att[h][j] * vp[j][tid];
    ctxs[tid] = c;
  }
  __syncthreads();
  float o = bo[tid];
  for (int k = 0; k < CCH; ++k) o += ctxs[k] * wo[(size_t)tid * CCH + k];
  y1[(size_t)n * CCH + tid] = vf[(size_t)n * CCH + tid] + o;
}

// --------------------- BatchNorm statistics (sum, sumsq) ---------------------
__global__ __launch_bounds__(256) void bn_stats_kernel(
    const float* __restrict__ src, int rows, float* __restrict__ sums) {
  __shared__ float s1[256], s2[256];
  const int tid = threadIdx.x;
  const int c   = tid & 63;
  float a = 0.f, b = 0.f;
  for (int row = blockIdx.x * 4 + (tid >> 6); row < rows; row += gridDim.x * 4) {
    const float v = src[(size_t)row * 64 + c];
    a += v; b += v * v;
  }
  s1[tid] = a; s2[tid] = b;
  __syncthreads();
  if (tid < 64) {
    a = s1[tid] + s1[tid + 64] + s1[tid + 128] + s1[tid + 192];
    b = s2[tid] + s2[tid + 64] + s2[tid + 128] + s2[tid + 192];
    atomicAdd(&sums[c], a);
    atomicAdd(&sums[64 + c], b);
  }
}

// ------------------- Kernel 4: BN1 + FFN (64->256->64) + residual ------------
__global__ __launch_bounds__(128) void ffn_kernel(
    const float* __restrict__ y1, const float* __restrict__ sums1,
    const float* __restrict__ g1, const float* __restrict__ b1,
    const __bf16* __restrict__ pl1, const float* __restrict__ l1b,
    const __bf16* __restrict__ pl2, const float* __restrict__ l2b,
    float* __restrict__ y2) {
  __shared__ __bf16 xb[16][CCH];
  __shared__ float  xf[16][CCH];
  __shared__ __bf16 hb[16][FF];
  const int tid = threadIdx.x;
  const int m0  = blockIdx.x * 16;
  const float invN = 1.f / (float)NVOX;
  for (int e = tid; e < 16 * CCH; e += 128) {
    const int r = e >> 6, c = e & 63;
    const float mu  = sums1[c] * invN;
    const float var = sums1[64 + c] * invN - mu * mu;
    const float rs  = rsqrtf(var + BN_EPS);
    const float x = (y1[(size_t)(m0 + r) * CCH + c] - mu) * rs * g1[c] + b1[c];
    xf[r][c] = x;
    xb[r][c] = (__bf16)x;
  }
  __syncthreads();
  const int w = tid >> 5, lane = tid & 31;
#pragma unroll
  for (int q = 0; q < 4; ++q) {             // each wave: 4 of 16 F-tiles
    const int ct = w * 4 + q;
    v8f acc = {};
#pragma unroll
    for (int ks = 0; ks < 2; ++ks) {
      v16bf a = frag_a_lds(&xb[0][0] + ks * 32, CCH);
      v16bf b = frag_b_packed(pl1, ct * 2 + ks);
      acc = wmma_bf16(a, b, acc);
    }
    const int f = ct * 16 + (lane & 15);
#pragma unroll
    for (int r = 0; r < 8; ++r) {
      const int m = ((lane >> 4) << 3) + r;
      hb[m][f] = (__bf16)fmaxf(acc[r] + l1b[f], 0.f);
    }
  }
  __syncthreads();
  {
    const int ct = w;                       // each wave: one C-tile of output
    v8f acc = {};
#pragma unroll
    for (int ks = 0; ks < 8; ++ks) {        // K = 256
      v16bf a = frag_a_lds(&hb[0][0] + ks * 32, FF);
      v16bf b = frag_b_packed(pl2, ct * 8 + ks);
      acc = wmma_bf16(a, b, acc);
    }
    const int c = ct * 16 + (lane & 15);
#pragma unroll
    for (int r = 0; r < 8; ++r) {
      const int m = ((lane >> 4) << 3) + r;
      y2[(size_t)(m0 + m) * CCH + c] = xf[m][c] + acc[r] + l2b[c];
    }
  }
}

// ------------------- Kernel 5: BN2 + output projection -----------------------
__global__ __launch_bounds__(32) void outproj_kernel(
    const float* __restrict__ y2, const float* __restrict__ sums2,
    const float* __restrict__ g2, const float* __restrict__ b2,
    const __bf16* __restrict__ pow_, const float* __restrict__ ob,
    float* __restrict__ zout) {
  __shared__ __bf16 xb[16][CCH];
  const int tid = threadIdx.x;
  const int m0  = blockIdx.x * 16;
  const float invN = 1.f / (float)NVOX;
  for (int e = tid; e < 16 * CCH; e += 32) {
    const int r = e >> 6, c = e & 63;
    const float mu  = sums2[c] * invN;
    const float var = sums2[64 + c] * invN - mu * mu;
    const float rs  = rsqrtf(var + BN_EPS);
    xb[r][c] = (__bf16)((y2[(size_t)(m0 + r) * CCH + c] - mu) * rs * g2[c] + b2[c]);
  }
  __syncthreads();
  const int lane = tid & 31;
#pragma unroll
  for (int ct = 0; ct < 4; ++ct) {
    v8f acc = {};
#pragma unroll
    for (int ks = 0; ks < 2; ++ks) {
      v16bf a = frag_a_lds(&xb[0][0] + ks * 32, CCH);
      v16bf b = frag_b_packed(pow_, ct * 2 + ks);
      acc = wmma_bf16(a, b, acc);
    }
    const int c = ct * 16 + (lane & 15);
#pragma unroll
    for (int r = 0; r < 8; ++r) {
      const int m = ((lane >> 4) << 3) + r;
      zout[(size_t)(m0 + m) * OCH + c] = acc[r] + ob[c];
    }
  }
}

// ------------------- Kernel 6: BN3 + relu -> output --------------------------
__global__ __launch_bounds__(256) void final_kernel(
    const float* __restrict__ z, const float* __restrict__ sums3,
    const float* __restrict__ g3, const float* __restrict__ b3,
    float* __restrict__ out) {
  const float invN = 1.f / (float)NVOX;
  for (size_t i = (size_t)blockIdx.x * 256 + threadIdx.x; i < (size_t)NVOX * OCH;
       i += (size_t)gridDim.x * 256) {
    const int c = (int)(i & 63);
    const float mu  = sums3[c] * invN;
    const float var = sums3[64 + c] * invN - mu * mu;
    const float rs  = rsqrtf(var + BN_EPS);
    out[i] = fmaxf((z[i] - mu) * rs * g3[c] + b3[c], 0.f);
  }
}

// -----------------------------------------------------------------------------
extern "C" void kernel_launch(void* const* d_in, const int* in_sizes, int n_in,
                              void* d_out, int out_size, void* d_ws, size_t ws_size,
                              hipStream_t stream) {
  const float* vf  = (const float*)d_in[0];
  const float* vc  = (const float*)d_in[1];
  const int*   ki  = (const int*)d_in[2];
  const unsigned char* km = (const unsigned char*)d_in[3];   // bool mask
  const float* wq  = (const float*)d_in[4];  const float* bq  = (const float*)d_in[5];
  const float* wk  = (const float*)d_in[6];  const float* bk  = (const float*)d_in[7];
  const float* wv  = (const float*)d_in[8];  const float* bv  = (const float*)d_in[9];
  const float* wo  = (const float*)d_in[10]; const float* bo  = (const float*)d_in[11];
  const float* qpw = (const float*)d_in[12]; const float* qpb = (const float*)d_in[13];
  const float* kpw = (const float*)d_in[14]; const float* kpb = (const float*)d_in[15];
  const float* g1  = (const float*)d_in[16]; const float* b1  = (const float*)d_in[17];
  const float* g2  = (const float*)d_in[18]; const float* b2  = (const float*)d_in[19];
  const float* l1w = (const float*)d_in[20]; const float* l1b = (const float*)d_in[21];
  const float* l2w = (const float*)d_in[22]; const float* l2b = (const float*)d_in[23];
  const float* ow  = (const float*)d_in[24]; const float* ob  = (const float*)d_in[25];
  const float* g3  = (const float*)d_in[26]; const float* b3  = (const float*)d_in[27];

  float* bufA  = (float*)d_ws;                    // q, then reused for y2
  float* bufB  = bufA + (size_t)NVOX * CCH;       // y1, then reused for z
  float* sums1 = bufB + (size_t)NVOX * CCH;       // [sum(64) | sumsq(64)] x3
  float* sums2 = sums1 + 128;
  float* sums3 = sums2 + 128;
  // pre-packed bf16 B fragments (each fragment = 32 lanes x 16 bf16 = 1KB)
  __bf16* pwq = (__bf16*)(sums3 + 128);            // 8 frags
  __bf16* pwk = pwq + 8 * 512;                     // 8 frags
  __bf16* pwv = pwk + 8 * 512;                     // 8 frags
  __bf16* pl1 = pwv + 8 * 512;                     // 32 frags (16 ct x 2 ks)
  __bf16* pl2 = pl1 + 32 * 512;                    // 32 frags (4 ct x 8 ks)
  __bf16* pow_ = pl2 + 32 * 512;                   // 8 frags

  hipMemsetAsync(sums1, 0, 3 * 128 * sizeof(float), stream);

  prepack_kernel<<<8,  32, 0, stream>>>(wq,  CCH, 2, pwq);
  prepack_kernel<<<8,  32, 0, stream>>>(wk,  CCH, 2, pwk);
  prepack_kernel<<<8,  32, 0, stream>>>(wv,  CCH, 2, pwv);
  prepack_kernel<<<32, 32, 0, stream>>>(l1w, CCH, 2, pl1);   // F x C
  prepack_kernel<<<32, 32, 0, stream>>>(l2w, FF,  8, pl2);   // C x F
  prepack_kernel<<<8,  32, 0, stream>>>(ow,  CCH, 2, pow_);

  const int nt = NVOX / 16;   // 6250
  qproj_kernel<<<nt, 32, 0, stream>>>(vf, vc, pwq, bq, qpw, qpb, bufA);
  attn_kernel<<<NVOX, 64, 0, stream>>>(vf, vc, ki, km, pwk, bk, pwv, bv, wo, bo,
                                       kpw, kpb, bufA, bufB);
  bn_stats_kernel<<<512, 256, 0, stream>>>(bufB, NVOX, sums1);
  ffn_kernel<<<nt, 128, 0, stream>>>(bufB, sums1, g1, b1, pl1, l1b, pl2, l2b, bufA);
  bn_stats_kernel<<<512, 256, 0, stream>>>(bufA, NVOX, sums2);
  outproj_kernel<<<nt, 32, 0, stream>>>(bufA, sums2, g2, b2, pow_, ob, bufB);
  bn_stats_kernel<<<512, 256, 0, stream>>>(bufB, NVOX, sums3);
  final_kernel<<<512, 256, 0, stream>>>(bufB, sums3, g3, b3, (float*)d_out);
}
